// AD_44006234915261
// MI455X (gfx1250) — compile-verified, matrix operand order checked by reading, not run on hardware
//
#include <hip/hip_runtime.h>
#include <hip/hip_bf16.h>

// ---------------- problem constants ----------------
#define BB   8
#define NN   511
#define LL   512            // n_transit
#define DD   1024
#define DFF  4096
#define HH   8
#define HD   128
#define NA   5
#define GRID9 9

typedef _Float16 h16;
typedef __attribute__((ext_vector_type(16))) _Float16 v16h;
typedef __attribute__((ext_vector_type(8)))  _Float16 v8h;
typedef __attribute__((ext_vector_type(8)))  float    v8f;

// GEMM epilogue modes
#define M_F32    0   // C = acc + bias                (float out)
#define M_SCALE  1   // C = acc * scale               (float out)
#define M_RES    2   // C = acc + bias + residual     (float out)
#define M_GELU16 3   // C = gelu(acc + bias)          (f16 out)
#define M_F16    4   // C = acc                       (f16 out)

// raw 32-bit LDS byte address (strip the generic-aperture bits via addrspace(3))
__device__ __forceinline__ unsigned lds_raw_addr(const void* p) {
  return (unsigned)(unsigned long long)(__attribute__((address_space(3))) const char*)p;
}

// async 16B global -> LDS (ASYNCcnt-tracked); offset imm applies to BOTH addresses
__device__ __forceinline__ void async16_0(unsigned lds, unsigned long long ga) {
  asm volatile("global_load_async_to_lds_b128 %0, %1, off" :: "v"(lds), "v"(ga) : "memory");
}
__device__ __forceinline__ void async16_16(unsigned lds, unsigned long long ga) {
  asm volatile("global_load_async_to_lds_b128 %0, %1, off offset:16" :: "v"(lds), "v"(ga) : "memory");
}
__device__ __forceinline__ void async16_32(unsigned lds, unsigned long long ga) {
  asm volatile("global_load_async_to_lds_b128 %0, %1, off offset:32" :: "v"(lds), "v"(ga) : "memory");
}
__device__ __forceinline__ void async16_48(unsigned lds, unsigned long long ga) {
  asm volatile("global_load_async_to_lds_b128 %0, %1, off offset:48" :: "v"(lds), "v"(ga) : "memory");
}
__device__ __forceinline__ void wait_async0() {
  asm volatile("s_wait_asynccnt 0x0" ::: "memory");
}

// ---------------------------------------------------------------------------
// Generic WMMA GEMM:  C[M,N] = A[M,K] (row-major, f16) x B[N,K]^T (row-major, f16)
// 256x128 tile per 256-thread block (8 waves, each wave = 64x64 = 4x4 frags).
// Tiles staged global->LDS with async-DMA, double-buffered (1 barrier / k-step).
// Inner loop: 16 ds_load_b128 : 16 WMMA.
// Batched via blockIdx.z; C batch offset = (z/CDIV)*cs1 + (z%CDIV)*cs2.
// ---------------------------------------------------------------------------
#define TLD 40   // padded LDS row stride in halfs (80B rows -> 16B-aligned b128)

template<int MODE, int CDIV>
__global__ __launch_bounds__(256) void gemm_wmma(
    const h16* __restrict__ A, const h16* __restrict__ B, void* __restrict__ Cv,
    const float* __restrict__ bias, const float* __restrict__ residual,
    long aBatch, long bBatch, long cs1, long cs2,
    int lda, int ldb, int ldc, int K, float scale)
{
  __shared__ h16 As[2][256 * TLD];   // 2 x 20 KB
  __shared__ h16 Bs[2][128 * TLD];   // 2 x 10 KB

  const int tid  = threadIdx.x;
  const int lane = tid & 31;
  const int wid  = tid >> 5;       // 0..7
  const int wm   = wid >> 1;       // 0..3  (64 rows each)
  const int wn   = wid & 1;        // 0..1  (64 cols each)
  const int z    = blockIdx.z;

  const long aOff = (long)z * aBatch;
  const long bOff = (long)z * bBatch;
  const long cOff = (long)(z / CDIV) * cs1 + (long)(z % CDIV) * cs2;

  const int tileM = blockIdx.x * 256;
  const int tileN = blockIdx.y * 128;

  v8f acc[4][4];
#pragma unroll
  for (int m = 0; m < 4; ++m)
#pragma unroll
    for (int n = 0; n < 4; ++n) acc[m][n] = (v8f)0.0f;

  const int fr = lane & 15;            // fragment row/col within 16
  const int kb = (lane >> 4) * 8;      // K sub-base per half-wave

  // staging: A = one full 32-half row per thread (4x b128),
  //          B = half row per thread (2x b128)
  const int bRow = tid >> 1;           // 0..127
  const int bCol = (tid & 1) * 16;     // 0 or 16 (halfs)
  const h16* aBase = A + aOff + (long)(tileM + tid) * lda;
  const h16* bBase = B + bOff + (long)(tileN + bRow) * ldb + bCol;
  const unsigned laA = lds_raw_addr(&As[0][tid * TLD]);
  const unsigned laB = lds_raw_addr(&Bs[0][bRow * TLD + bCol]);
  const unsigned strideA = 256 * TLD * sizeof(h16);   // 20480 B
  const unsigned strideB = 128 * TLD * sizeof(h16);   // 10240 B

  auto issue_tile = [&](int buf, int k0) {
    const unsigned long long ga = (unsigned long long)(aBase + k0);
    const unsigned la = laA + (unsigned)buf * strideA;
    async16_0 (la, ga);
    async16_16(la, ga);
    async16_32(la, ga);
    async16_48(la, ga);
    const unsigned long long gb = (unsigned long long)(bBase + k0);
    const unsigned lb = laB + (unsigned)buf * strideB;
    async16_0 (lb, gb);
    async16_16(lb, gb);
  };

  issue_tile(0, 0);

  for (int k0 = 0; k0 < K; k0 += 32) {
    const int cur = (k0 >> 5) & 1;
    wait_async0();          // own async loads done
    __syncthreads();        // everyone's loads done; prev reads of other buf done
    if (k0 + 32 < K) issue_tile(cur ^ 1, k0 + 32);

    // ---- load fragments, 16 WMMAs per k-step
    v16h bf[4];
#pragma unroll
    for (int n = 0; n < 4; ++n) {
      const int br = wn * 64 + n * 16 + fr;
      v8h lo = *(const v8h*)&Bs[cur][br * TLD + kb];
      v8h hi = *(const v8h*)&Bs[cur][br * TLD + kb + 16];
      bf[n] = __builtin_shufflevector(lo, hi, 0,1,2,3,4,5,6,7,8,9,10,11,12,13,14,15);
    }
#pragma unroll
    for (int m = 0; m < 4; ++m) {
      const int ar = wm * 64 + m * 16 + fr;
      v8h lo = *(const v8h*)&As[cur][ar * TLD + kb];
      v8h hi = *(const v8h*)&As[cur][ar * TLD + kb + 16];
      v16h af = __builtin_shufflevector(lo, hi, 0,1,2,3,4,5,6,7,8,9,10,11,12,13,14,15);
#pragma unroll
      for (int n = 0; n < 4; ++n)
        acc[m][n] = __builtin_amdgcn_wmma_f32_16x16x32_f16(
            false, af, false, bf[n], (short)0, acc[m][n], false, false);
    }
  }

  // ---- epilogue
  const int rowAdd = (lane >> 4) * 8;
#pragma unroll
  for (int m = 0; m < 4; ++m) {
#pragma unroll
    for (int n = 0; n < 4; ++n) {
      const int gc = tileN + wn * 64 + n * 16 + fr;
      float bcol = 0.0f;
      if (MODE == M_F32 || MODE == M_RES || MODE == M_GELU16) bcol = bias[gc];
#pragma unroll
      for (int r = 0; r < 8; ++r) {
        const int gr = tileM + wm * 64 + m * 16 + rowAdd + r;
        const long ci = cOff + (long)gr * ldc + gc;
        const float v = acc[m][n][r];
        if (MODE == M_F32) {
          ((float*)Cv)[ci] = v + bcol;
        } else if (MODE == M_SCALE) {
          ((float*)Cv)[ci] = v * scale;
        } else if (MODE == M_RES) {
          ((float*)Cv)[ci] = v + bcol + residual[ci];
        } else if (MODE == M_GELU16) {
          float t = v + bcol;
          t = 0.5f * t * (1.0f + erff(t * 0.70710678118654752f));
          ((h16*)Cv)[ci] = (h16)t;
        } else { // M_F16
          ((h16*)Cv)[ci] = (h16)v;
        }
      }
    }
  }
}

// ---------------------------------------------------------------------------
// Elementwise / reduction kernels
// ---------------------------------------------------------------------------
__global__ void f32_to_f16_kernel(const float* __restrict__ in, h16* __restrict__ out, long n)
{
  long i = (long)blockIdx.x * 256 + threadIdx.x;
  if (i < n) out[i] = (h16)in[i];
}

__global__ void embed_kernel(const float* __restrict__ states, const float* __restrict__ actions,
                             const float* __restrict__ next_states, const float* __restrict__ rewards,
                             const float* __restrict__ pos, const float* __restrict__ Wc,
                             const float* __restrict__ bc, const float* __restrict__ qtable,
                             const int* __restrict__ qstates,
                             float* __restrict__ x, h16* __restrict__ x16)
{
  long i = (long)blockIdx.x * 256 + threadIdx.x;       // over B*L*D
  if (i >= (long)BB * LL * DD) return;
  int  d  = (int)(i % DD);
  long bl = i / DD;
  int  l  = (int)(bl % LL);
  int  b  = (int)(bl / LL);
  float v;
  if (l < NN) {
    const float* st = states      + ((long)b * NN + l) * 2;
    const float* ac = actions     + ((long)b * NN + l) * NA;
    const float* ns = next_states + ((long)b * NN + l) * 2;
    float rw = rewards[(long)b * NN + l];
    const float* w = Wc + (long)d * 10;
    v = bc[d]
      + w[0]*st[0] + w[1]*st[1]
      + w[2]*ac[0] + w[3]*ac[1] + w[4]*ac[2] + w[5]*ac[3] + w[6]*ac[4]
      + w[7]*rw
      + w[8]*ns[0] + w[9]*ns[1];
  } else {
    int qi = qstates[b * 2] * GRID9 + qstates[b * 2 + 1];
    v = qtable[(long)qi * DD + d];
  }
  v += pos[(long)l * DD + d];
  x[i]   = v;
  x16[i] = (h16)v;
}

__global__ void split_qkv_kernel(const float* __restrict__ qkv,
                                 h16* __restrict__ q, h16* __restrict__ k, h16* __restrict__ vT)
{
  long i = (long)blockIdx.x * 256 + threadIdx.x;       // over B*L*3D
  if (i >= (long)BB * LL * 3 * DD) return;
  int  c3 = (int)(i % (3 * DD));
  long bl = i / (3 * DD);
  int  l  = (int)(bl % LL);
  int  b  = (int)(bl / LL);
  int  part = c3 / DD;
  int  d  = c3 % DD;
  int  h  = d / HD;
  int  c  = d % HD;
  h16 val = (h16)qkv[i];
  long zo = (long)(b * HH + h);
  if (part == 0)      q [(zo * LL + l) * HD + c] = val;
  else if (part == 1) k [(zo * LL + l) * HD + c] = val;
  else                vT[(zo * HD + c) * LL + l] = val;
}

__global__ __launch_bounds__(256) void softmax_kernel(float* __restrict__ attn, h16* __restrict__ a16)
{
  __shared__ float red[256];
  const long row = blockIdx.x;
  float* p = attn + row * (long)LL;
  const int tid = threadIdx.x;
  float v0 = p[tid], v1 = p[tid + 256];
  red[tid] = fmaxf(v0, v1);
  __syncthreads();
  for (int s = 128; s > 0; s >>= 1) { if (tid < s) red[tid] = fmaxf(red[tid], red[tid + s]); __syncthreads(); }
  const float mx = red[0];
  __syncthreads();
  float e0 = expf(v0 - mx), e1 = expf(v1 - mx);
  red[tid] = e0 + e1;
  __syncthreads();
  for (int s = 128; s > 0; s >>= 1) { if (tid < s) red[tid] += red[tid + s]; __syncthreads(); }
  const float inv = 1.0f / red[0];
  const float o0 = e0 * inv, o1 = e1 * inv;
  p[tid]       = o0;  p[tid + 256]       = o1;
  a16[row * (long)LL + tid]       = (h16)o0;
  a16[row * (long)LL + tid + 256] = (h16)o1;
}

__global__ __launch_bounds__(256) void ln_kernel(const float* __restrict__ in,
                                                 const float* __restrict__ g, const float* __restrict__ bp,
                                                 float* __restrict__ xo, h16* __restrict__ xo16)
{
  __shared__ float red[256];
  const long row = blockIdx.x;
  const float* p = in + row * (long)DD;
  const int tid = threadIdx.x;
  float vals[4];
  float s = 0.0f, ss = 0.0f;
#pragma unroll
  for (int j = 0; j < 4; ++j) {
    float v = p[tid + j * 256];
    vals[j] = v; s += v; ss += v * v;
  }
  red[tid] = s; __syncthreads();
  for (int st = 128; st > 0; st >>= 1) { if (tid < st) red[tid] += red[tid + st]; __syncthreads(); }
  s = red[0]; __syncthreads();
  red[tid] = ss; __syncthreads();
  for (int st = 128; st > 0; st >>= 1) { if (tid < st) red[tid] += red[tid + st]; __syncthreads(); }
  ss = red[0];
  const float mean = s * (1.0f / DD);
  const float var  = ss * (1.0f / DD) - mean * mean;
  const float rstd = rsqrtf(var + 1e-5f);
#pragma unroll
  for (int j = 0; j < 4; ++j) {
    const int d = tid + j * 256;
    const float v = (vals[j] - mean) * rstd * g[d] + bp[d];
    xo  [row * (long)DD + d] = v;
    xo16[row * (long)DD + d] = (h16)v;
  }
}

__global__ __launch_bounds__(256) void head_kernel(const float* __restrict__ x,
                                                   const float* __restrict__ pw, const float* __restrict__ pb,
                                                   float* __restrict__ logits)
{
  __shared__ float red[256];
  const int b = blockIdx.x, tid = threadIdx.x;
  const float* xr = x + ((long)b * LL + (LL - 1)) * DD;
  for (int a = 0; a < NA; ++a) {
    float s = 0.0f;
    for (int d = tid; d < DD; d += 256) s += xr[d] * pw[a * DD + d];
    red[tid] = s; __syncthreads();
    for (int st = 128; st > 0; st >>= 1) { if (tid < st) red[tid] += red[tid + st]; __syncthreads(); }
    if (tid == 0) logits[b * NA + a] = red[0] + pb[a];
    __syncthreads();
  }
}

__global__ void loss_kernel(const float* __restrict__ logits, const int* __restrict__ target,
                            float* __restrict__ out)
{
  if (threadIdx.x != 0 || blockIdx.x != 0) return;
  float loss = 0.0f, acc = 0.0f;
  for (int b = 0; b < BB; ++b) {
    const float* l = logits + b * NA;
    float m = l[0]; int am = 0;
    for (int a = 1; a < NA; ++a) if (l[a] > m) { m = l[a]; am = a; }
    float se = 0.0f;
    for (int a = 0; a < NA; ++a) se += expf(l[a] - m);
    const float lse = m + logf(se);
    float slp = 0.0f;
    for (int a = 0; a < NA; ++a) slp += l[a] - lse;
    const int t = target[b];
    const float nll = -(l[t] - lse);
    loss += (1.0f - 0.1f) * nll - 0.1f * (slp * (1.0f / NA));
    acc  += (am == t) ? 1.0f : 0.0f;
  }
  out[0] = loss * (1.0f / BB);
  out[1] = acc  * (1.0f / BB);
}

// ---------------------------------------------------------------------------
extern "C" void kernel_launch(void* const* d_in, const int* in_sizes, int n_in,
                              void* d_out, int out_size, void* d_ws, size_t ws_size,
                              hipStream_t stream)
{
  const float* states      = (const float*)d_in[0];
  const float* actions     = (const float*)d_in[1];
  const float* next_states = (const float*)d_in[2];
  const float* rewards     = (const float*)d_in[3];
  const float* pos_emb     = (const float*)d_in[4];
  const float* emb_ctx_w   = (const float*)d_in[5];
  const float* emb_ctx_b   = (const float*)d_in[6];
  const float* emb_q_tab   = (const float*)d_in[7];
  const float* in_proj_w   = (const float*)d_in[8];
  const float* in_proj_b   = (const float*)d_in[9];
  const float* out_proj_w  = (const float*)d_in[10];
  const float* out_proj_b  = (const float*)d_in[11];
  const float* linear1_w   = (const float*)d_in[12];
  const float* linear1_b   = (const float*)d_in[13];
  const float* linear2_w   = (const float*)d_in[14];
  const float* linear2_b   = (const float*)d_in[15];
  const float* norm1_g     = (const float*)d_in[16];
  const float* norm1_b     = (const float*)d_in[17];
  const float* norm2_g     = (const float*)d_in[18];
  const float* norm2_b     = (const float*)d_in[19];
  const float* pred_w      = (const float*)d_in[20];
  const float* pred_b      = (const float*)d_in[21];
  const int*   query_st    = (const int*)d_in[22];
  const int*   target_act  = (const int*)d_in[23];

  const long BL  = (long)BB * LL;          // 4096
  const long BHL = (long)BB * HH * LL;     // 32768
  const long ATT = (long)BB * HH * LL * LL;

  char* ws = (char*)d_ws;
  size_t off = 0;
  auto carve = [&](size_t bytes) -> char* {
    char* p = ws + off;
    off = (off + bytes + 255) & ~(size_t)255;
    return p;
  };

  float* x      = (float*)carve(BL * DD * 4);
  h16*   x16    = (h16*)  carve(BL * DD * 2);
  float* t0     = (float*)carve(BL * 3 * DD * 4);     // qkv / residual-out scratch
  h16*   q16    = (h16*)  carve(BHL * HD * 2);
  h16*   k16    = (h16*)  carve(BHL * HD * 2);
  h16*   v16t   = (h16*)  carve(BHL * HD * 2);
  h16*   attn16 = (h16*)  carve(ATT * 2);
  h16*   ctx16  = (h16*)  carve(BL * DD * 2);
  h16*   ff16   = (h16*)  carve(BL * DFF * 2);
  h16*   win16  = (h16*)  carve((size_t)3 * DD * DD * 2);
  h16*   wout16 = (h16*)  carve((size_t)DD * DD * 2);
  h16*   w1_16  = (h16*)  carve((size_t)DFF * DD * 2);
  h16*   w2_16  = (h16*)  carve((size_t)DD * DFF * 2);
  float* logits = (float*)carve(64 * 4);

  auto cvt = [&](const float* src, h16* dst, long n) {
    f32_to_f16_kernel<<<(int)((n + 255) / 256), 256, 0, stream>>>(src, dst, n);
  };
  cvt(in_proj_w,  win16,  (long)3 * DD * DD);
  cvt(out_proj_w, wout16, (long)DD * DD);
  cvt(linear1_w,  w1_16,  (long)DFF * DD);
  cvt(linear2_w,  w2_16,  (long)DD * DFF);

  embed_kernel<<<(int)((BL * DD + 255) / 256), 256, 0, stream>>>(
      states, actions, next_states, rewards, pos_emb, emb_ctx_w, emb_ctx_b,
      emb_q_tab, query_st, x, x16);

  const float inv_sqrt_hd = 0.08838834764831845f;  // 1/sqrt(128)

  for (int layer = 0; layer < 4; ++layer) {
    float* attnL = (float*)d_out + 2 + (size_t)layer * ATT;

    // QKV: (4096x1024) x (3072x1024)^T -> t0 (f32 + bias)
    gemm_wmma<M_F32, 1><<<dim3(16, 24, 1), 256, 0, stream>>>(
        x16, win16, t0, in_proj_b, nullptr,
        0, 0, 0, 0, DD, DD, 3 * DD, DD, 1.0f);

    split_qkv_kernel<<<(int)((BL * 3 * DD + 255) / 256), 256, 0, stream>>>(
        t0, q16, k16, v16t);

    // scores: per (b,h): Q(512x128) x K(512x128)^T * scale -> d_out attn slab
    gemm_wmma<M_SCALE, 1><<<dim3(2, 4, BB * HH), 256, 0, stream>>>(
        q16, k16, attnL, nullptr, nullptr,
        (long)LL * HD, (long)LL * HD, (long)LL * LL, 0,
        HD, HD, LL, HD, inv_sqrt_hd);

    softmax_kernel<<<(int)BHL, 256, 0, stream>>>(attnL, attn16);

    // ctx: per (b,h): attn(512x512) x V^T(128x512)^T -> ctx16 (b,l,h*128+c)
    gemm_wmma<M_F16, HH><<<dim3(2, 1, BB * HH), 256, 0, stream>>>(
        attn16, v16t, ctx16, nullptr, nullptr,
        (long)LL * LL, (long)HD * LL, (long)LL * DD, (long)HD,
        LL, LL, DD, LL, 1.0f);

    // out_proj + residual -> t0 ; LN1 -> x, x16
    gemm_wmma<M_RES, 1><<<dim3(16, 8, 1), 256, 0, stream>>>(
        ctx16, wout16, t0, out_proj_b, x,
        0, 0, 0, 0, DD, DD, DD, DD, 1.0f);
    ln_kernel<<<(int)BL, 256, 0, stream>>>(t0, norm1_g, norm1_b, x, x16);

    // FFN: gelu(x @ W1^T + b1) -> ff16 ; W2 + residual -> t0 ; LN2 -> x, x16
    gemm_wmma<M_GELU16, 1><<<dim3(16, 32, 1), 256, 0, stream>>>(
        x16, w1_16, ff16, linear1_b, nullptr,
        0, 0, 0, 0, DD, DD, DFF, DD, 1.0f);
    gemm_wmma<M_RES, 1><<<dim3(16, 8, 1), 256, 0, stream>>>(
        ff16, w2_16, t0, linear2_b, x,
        0, 0, 0, 0, DFF, DFF, DD, DFF, 1.0f);
    ln_kernel<<<(int)BL, 256, 0, stream>>>(t0, norm2_g, norm2_b, x, x16);
  }

  head_kernel<<<BB, 256, 0, stream>>>(x, pred_w, pred_b, logits);
  loss_kernel<<<1, 32, 0, stream>>>(logits, target_act, (float*)d_out);
}
